// MambaBlock_31215822307422
// MI455X (gfx1250) — matrix operand order repaired and checked by reading, not compiled, not run on hardware
//
#include <hip/hip_runtime.h>
#include <hip/hip_bf16.h>

// ---------------- problem constants ----------------
#define BB      2
#define TT      1024
#define DM      768        // d_model
#define DS      64         // d_state
#define DC      4          // d_conv
#define DTR     32         // dt_rank
#define DI      1536       // d_inner
#define MROWS   (BB*TT)    // 2048
#define XCOLS   (DTR + 2*DS)   // 160

// ---------------- WMMA types (gfx1250, wave32) ----------------
typedef __attribute__((ext_vector_type(16))) __bf16 v16bf;
typedef __attribute__((ext_vector_type(8)))  float  v8f;

// float -> bf16 (round to nearest even), stored as u16
__device__ __forceinline__ unsigned short f2bf(float f) {
    unsigned int u = __float_as_uint(f);
    if ((u & 0x7F800000u) == 0x7F800000u) return (unsigned short)(u >> 16); // inf/nan
    u += 0x7FFFu + ((u >> 16) & 1u);
    return (unsigned short)(u >> 16);
}

__device__ __forceinline__ float siluf(float v) { return v / (1.f + __expf(-v)); }

// ---------------- RMSNorm -> bf16 ----------------
__global__ __launch_bounds__(256) void rmsnorm_bf16_kernel(
    const float* __restrict__ x, const float* __restrict__ w,
    unsigned short* __restrict__ out) {
    __shared__ float sdata[256];
    const int row = blockIdx.x;                    // 0..MROWS-1
    const float* xr = x + (size_t)row * DM;
    float s = 0.f;
    for (int i = threadIdx.x; i < DM; i += 256) { float v = xr[i]; s += v * v; }
    sdata[threadIdx.x] = s;
    __syncthreads();
    for (int st = 128; st > 0; st >>= 1) {
        if (threadIdx.x < st) sdata[threadIdx.x] += sdata[threadIdx.x + st];
        __syncthreads();
    }
    const float scale = rsqrtf(sdata[0] * (1.0f / DM) + 1e-6f);
    for (int i = threadIdx.x; i < DM; i += 256)
        out[(size_t)row * DM + i] = f2bf(xr[i] * scale * w[i]);
}

// ---------------- transpose + cast: in f32 [K,N] -> out bf16 [N,K] ----------------
__global__ __launch_bounds__(256) void transcast_kernel(
    const float* __restrict__ in, unsigned short* __restrict__ out,
    int K, int N, int total) {
    int i = blockIdx.x * 256 + threadIdx.x;
    if (i >= total) return;
    int n = i / K, k = i - n * K;
    out[i] = f2bf(in[(size_t)k * N + n]);
}

// ---------------- bf16 WMMA GEMM: C[M,N] = A[M,K] * Bt[N,K]^T (+resid) ----------------
// Wave tile: 32x32 (2x2 WMMA register tiles). Block = 8 waves = 128(M) x 64(N).
// Requires M%128==0, K%32==0. N handled with clamped loads + guarded stores.
union FragAB { v16bf v; uint4 q[2]; };
union FragC  { v8f  v; float f[8]; };

__device__ __forceinline__ void store_tile(
    const v8f& acc, float* __restrict__ C, const float* __restrict__ resid,
    int rowBase, int colBase, int r, int hi, int N) {
    const int col = colBase + r;
    if (col >= N) return;
    FragC fc; fc.v = acc;
    const int row0 = rowBase + hi * 8;   // VGPR i -> row rowBase + i + hi*8
    #pragma unroll
    for (int i = 0; i < 8; ++i) {
        const size_t idx = (size_t)(row0 + i) * N + col;
        float v = fc.f[i];
        if (resid) v += resid[idx];
        C[idx] = v;
    }
}

__global__ __launch_bounds__(256) void gemm_bf16_wmma_kernel(
    const unsigned short* __restrict__ A,   // [M,K] bf16 row-major
    const unsigned short* __restrict__ Bt,  // [N,K] bf16 row-major (B transposed)
    float* __restrict__ C, int M, int N, int K,
    const float* __restrict__ resid) {
    const int lane = threadIdx.x & 31;
    const int wave = threadIdx.x >> 5;
    const int mRow = blockIdx.y * 128 + (wave >> 1) * 32;
    const int nCol = blockIdx.x * 64  + (wave & 1) * 32;
    const int r  = lane & 15;
    const int hi = lane >> 4;   // lane half: selects K sub-blocks / M+8 rows

    // A fragments (16x32 bf16): lane half hi -> K offsets {hi*8+0..7, 16+hi*8+0..7}
    const unsigned short* a0 = A + (size_t)(mRow + r) * K + hi * 8;
    const unsigned short* a1 = a0 + (size_t)16 * K;
    // B fragments (32x16 bf16): column fixed per lane, K offsets {hi*8+0..7, 16+hi*8+0..7}.
    // Clamp columns so addresses stay in-bounds when N % 64 != 0 (stores are guarded).
    int c0 = nCol + r;      if (c0 > N - 1) c0 = N - 1;
    int c1 = nCol + 16 + r; if (c1 > N - 1) c1 = N - 1;
    const unsigned short* b0 = Bt + (size_t)c0 * K + hi * 8;
    const unsigned short* b1 = Bt + (size_t)c1 * K + hi * 8;

    v8f acc00 = {}, acc01 = {}, acc10 = {}, acc11 = {};
    for (int kk = 0; kk < K; kk += 32) {
        FragAB fa0, fa1, fb0, fb1;
        fa0.q[0] = *reinterpret_cast<const uint4*>(a0 + kk);
        fa0.q[1] = *reinterpret_cast<const uint4*>(a0 + kk + 16);
        fa1.q[0] = *reinterpret_cast<const uint4*>(a1 + kk);
        fa1.q[1] = *reinterpret_cast<const uint4*>(a1 + kk + 16);
        fb0.q[0] = *reinterpret_cast<const uint4*>(b0 + kk);
        fb0.q[1] = *reinterpret_cast<const uint4*>(b0 + kk + 16);
        fb1.q[0] = *reinterpret_cast<const uint4*>(b1 + kk);
        fb1.q[1] = *reinterpret_cast<const uint4*>(b1 + kk + 16);
        // speculative prefetch of next K-step (safe to over-run allocation)
        __builtin_prefetch(a0 + kk + 32, 0, 1);
        __builtin_prefetch(a1 + kk + 32, 0, 1);
        __builtin_prefetch(b0 + kk + 32, 0, 1);
        __builtin_prefetch(b1 + kk + 32, 0, 1);
        acc00 = __builtin_amdgcn_wmma_f32_16x16x32_bf16(false, fa0.v, false, fb0.v, (short)0, acc00, false, false);
        acc01 = __builtin_amdgcn_wmma_f32_16x16x32_bf16(false, fa0.v, false, fb1.v, (short)0, acc01, false, false);
        acc10 = __builtin_amdgcn_wmma_f32_16x16x32_bf16(false, fa1.v, false, fb0.v, (short)0, acc10, false, false);
        acc11 = __builtin_amdgcn_wmma_f32_16x16x32_bf16(false, fa1.v, false, fb1.v, (short)0, acc11, false, false);
    }

    store_tile(acc00, C, resid, mRow,      nCol,      r, hi, N);
    store_tile(acc01, C, resid, mRow,      nCol + 16, r, hi, N);
    store_tile(acc10, C, resid, mRow + 16, nCol,      r, hi, N);
    store_tile(acc11, C, resid, mRow + 16, nCol + 16, r, hi, N);
}

// ---------------- causal depthwise conv1d + SiLU ----------------
// x_ssm = silu(conv(xz[..., :DI]) + bias); writes f32 (for scan) and bf16 (for GEMM2)
__global__ __launch_bounds__(256) void conv_silu_kernel(
    const float* __restrict__ xz, const float* __restrict__ cw,
    const float* __restrict__ cb, float* __restrict__ xf,
    unsigned short* __restrict__ xbf, int total) {
    int i = blockIdx.x * 256 + threadIdx.x;
    if (i >= total) return;
    const int d = i % DI;
    const int m = i / DI;          // b*TT + t
    const int t = m % TT;
    const int b = m / TT;
    float acc = cb[d];
    #pragma unroll
    for (int j = 0; j < DC; ++j) {
        const int tt = t - (DC - 1) + j;
        if (tt >= 0)
            acc += cw[j * DI + d] * xz[((size_t)(b * TT + tt)) * (2 * DI) + d];
    }
    const float s = siluf(acc);
    xf[i]  = s;
    xbf[i] = f2bf(s);
}

// ---------------- extract dt_lo (first DTR cols of x_dbl) as bf16 ----------------
__global__ __launch_bounds__(256) void extract_dtlo_kernel(
    const float* __restrict__ xdbl, unsigned short* __restrict__ out, int total) {
    int i = blockIdx.x * 256 + threadIdx.x;
    if (i >= total) return;
    const int m = i / DTR, c = i - m * DTR;
    out[i] = f2bf(xdbl[(size_t)m * XCOLS + c]);
}

// ---------------- softplus(dt_pre + b_dt), in place ----------------
__global__ __launch_bounds__(256) void softplus_kernel(
    float* __restrict__ dt, const float* __restrict__ bdt, int total) {
    int i = blockIdx.x * 256 + threadIdx.x;
    if (i >= total) return;
    const float v = dt[i] + bdt[i % DI];
    dt[i] = (v > 20.f) ? v : log1pf(__expf(v));
}

// ---------------- selective scan (sequential over T) ----------------
// One thread owns one (b, d) channel, keeps h[64] and A-row[64] in VGPRs.
// B_t / C_t broadcast through LDS each step.
__global__ __launch_bounds__(256) void scan_kernel(
    const float* __restrict__ xdbl, const float* __restrict__ dt,
    const float* __restrict__ xssm, const float* __restrict__ Alog,
    float* __restrict__ ys) {
    __shared__ float Bs[DS];
    __shared__ float Cs[DS];
    const int b = blockIdx.y;
    const int d = blockIdx.x * 256 + threadIdx.x;   // < DI by construction

    float Arow[DS], h[DS];
    #pragma unroll
    for (int n = 0; n < DS; ++n) {
        Arow[n] = -__expf(Alog[(size_t)d * DS + n]);
        h[n] = 0.f;
    }

    for (int t = 0; t < TT; ++t) {
        const size_t m = (size_t)b * TT + t;
        if (threadIdx.x < DS)
            Bs[threadIdx.x] = xdbl[m * XCOLS + DTR + threadIdx.x];
        else if (threadIdx.x < 2 * DS)
            Cs[threadIdx.x - DS] = xdbl[m * XCOLS + DTR + DS + (threadIdx.x - DS)];
        __syncthreads();

        const float dtv = dt[m * DI + d];
        const float xv  = xssm[m * DI + d];
        const float dBx = dtv * xv;
        float y = 0.f;
        #pragma unroll
        for (int n = 0; n < DS; ++n) {
            const float dA = __expf(dtv * Arow[n]);
            h[n] = dA * h[n] + dBx * Bs[n];
            y += h[n] * Cs[n];
        }
        ys[m * DI + d] = y;
        __syncthreads();   // protect Bs/Cs before next step overwrites
    }
}

// ---------------- y = (ys + x_ssm*D) * silu(z) -> bf16 ----------------
__global__ __launch_bounds__(256) void gate_kernel(
    const float* __restrict__ ys, const float* __restrict__ xssm,
    const float* __restrict__ Dp, const float* __restrict__ xz,
    unsigned short* __restrict__ ybf, int total) {
    int i = blockIdx.x * 256 + threadIdx.x;
    if (i >= total) return;
    const int d = i % DI;
    const size_t m = (size_t)(i / DI);
    const float y = ys[i] + xssm[i] * Dp[d];
    const float z = xz[m * (2 * DI) + DI + d];
    ybf[i] = f2bf(y * siluf(z));
}

// ---------------- host-side pipeline ----------------
extern "C" void kernel_launch(void* const* d_in, const int* in_sizes, int n_in,
                              void* d_out, int out_size, void* d_ws, size_t ws_size,
                              hipStream_t stream) {
    const float* x       = (const float*)d_in[0];   // [B,T,DM]
    const float* norm_w  = (const float*)d_in[1];   // [DM]
    const float* W_in    = (const float*)d_in[2];   // [DM, 2*DI]
    const float* conv_w  = (const float*)d_in[3];   // [DC,1,DI]
    const float* conv_b  = (const float*)d_in[4];   // [DI]
    const float* W_x     = (const float*)d_in[5];   // [DI, XCOLS]
    const float* W_dt    = (const float*)d_in[6];   // [DTR, DI]
    const float* b_dt    = (const float*)d_in[7];   // [DI]
    const float* A_log   = (const float*)d_in[8];   // [DI, DS]
    const float* D_param = (const float*)d_in[9];   // [DI]
    const float* W_out   = (const float*)d_in[10];  // [DI, DM]
    float* out = (float*)d_out;                     // [B,T,DM]

    // bump allocator over d_ws
    size_t off = 0;
    auto alloc = [&](size_t bytes) -> void* {
        off = (off + 255) & ~size_t(255);
        void* p = (char*)d_ws + off;
        off += bytes;
        return p;
    };
    unsigned short* xn_bf   = (unsigned short*)alloc((size_t)MROWS * DM   * 2);
    unsigned short* WinT    = (unsigned short*)alloc((size_t)(2*DI) * DM  * 2);
    unsigned short* WxT     = (unsigned short*)alloc((size_t)XCOLS * DI   * 2);
    unsigned short* WdtT    = (unsigned short*)alloc((size_t)DI    * DTR  * 2);
    unsigned short* WoutT   = (unsigned short*)alloc((size_t)DM    * DI   * 2);
    float*          xz      = (float*)alloc((size_t)MROWS * 2 * DI * 4);
    float*          xssm_f  = (float*)alloc((size_t)MROWS * DI    * 4);
    unsigned short* xssm_bf = (unsigned short*)alloc((size_t)MROWS * DI * 2);
    float*          xdbl    = (float*)alloc((size_t)MROWS * XCOLS * 4);
    unsigned short* dtlo_bf = (unsigned short*)alloc((size_t)MROWS * DTR * 2);
    float*          dtbuf   = (float*)alloc((size_t)MROWS * DI * 4);  // dt_pre -> dt
    float*          ysbuf   = (float*)alloc((size_t)MROWS * DI * 4);
    unsigned short* y_bf    = (unsigned short*)alloc((size_t)MROWS * DI * 2);
    (void)ws_size; (void)in_sizes; (void)n_in; (void)out_size;

    const int EB = 256;
    auto ceil_b = [](int n) { return (n + 255) / 256; };
    auto gemm_grid = [](int Mv, int Nv) { return dim3((Nv + 63) / 64, Mv / 128); };

    // 1) RMSNorm -> bf16
    rmsnorm_bf16_kernel<<<MROWS, EB, 0, stream>>>(x, norm_w, xn_bf);

    // 2) weight cast+transpose (bf16, [N,K])
    transcast_kernel<<<ceil_b(2*DI*DM), EB, 0, stream>>>(W_in,  WinT,  DM,  2*DI, 2*DI*DM);
    transcast_kernel<<<ceil_b(XCOLS*DI), EB, 0, stream>>>(W_x,  WxT,   DI,  XCOLS, XCOLS*DI);
    transcast_kernel<<<ceil_b(DI*DTR),  EB, 0, stream>>>(W_dt,  WdtT,  DTR, DI,   DI*DTR);
    transcast_kernel<<<ceil_b(DM*DI),   EB, 0, stream>>>(W_out, WoutT, DI,  DM,   DM*DI);

    // 3) xz = xn @ W_in    [2048 x 3072], K=768
    gemm_bf16_wmma_kernel<<<gemm_grid(MROWS, 2*DI), EB, 0, stream>>>(
        xn_bf, WinT, xz, MROWS, 2*DI, DM, nullptr);

    // 4) causal depthwise conv + SiLU
    conv_silu_kernel<<<ceil_b(MROWS*DI), EB, 0, stream>>>(
        xz, conv_w, conv_b, xssm_f, xssm_bf, MROWS*DI);

    // 5) x_dbl = x_ssm @ W_x    [2048 x 160], K=1536
    gemm_bf16_wmma_kernel<<<gemm_grid(MROWS, XCOLS), EB, 0, stream>>>(
        xssm_bf, WxT, xdbl, MROWS, XCOLS, DI, nullptr);

    // 6) dt = softplus(dt_lo @ W_dt + b_dt)   [2048 x 1536], K=32
    extract_dtlo_kernel<<<ceil_b(MROWS*DTR), EB, 0, stream>>>(xdbl, dtlo_bf, MROWS*DTR);
    gemm_bf16_wmma_kernel<<<gemm_grid(MROWS, DI), EB, 0, stream>>>(
        dtlo_bf, WdtT, dtbuf, MROWS, DI, DTR, nullptr);
    softplus_kernel<<<ceil_b(MROWS*DI), EB, 0, stream>>>(dtbuf, b_dt, MROWS*DI);

    // 7) selective scan (sequential over T, parallel over B x DI)
    scan_kernel<<<dim3(DI/256, BB), EB, 0, stream>>>(xdbl, dtbuf, xssm_f, A_log, ysbuf);

    // 8) gating: y = (ys + x_ssm*D) * silu(z) -> bf16
    gate_kernel<<<ceil_b(MROWS*DI), EB, 0, stream>>>(
        ysbuf, xssm_f, D_param, xz, y_bf, MROWS*DI);

    // 9) out = residual + y @ W_out   [2048 x 768], K=1536, fused residual add
    gemm_bf16_wmma_kernel<<<gemm_grid(MROWS, DM), EB, 0, stream>>>(
        y_bf, WoutT, out, MROWS, DM, DI, x);
}
